// CNV_65549790871624
// MI455X (gfx1250) — 8-bit matrix operand order repaired and checked by reading, compiled, not run on hardware
//
#include <hip/hip_runtime.h>
#include <stdint.h>


typedef int v8i __attribute__((ext_vector_type(8)));

#define BN_EPS 1e-4f

__global__ __launch_bounds__(256) void k_quant(const float* __restrict__ x,
                                               int8_t* __restrict__ out) {
  int i = blockIdx.x * 256 + threadIdx.x;
  if (i >= 256 * 1024) return;
  int n = i >> 10, p = i & 1023;
  int packed = 0;
  for (int c = 0; c < 3; ++c) {
    float h = 2.f * x[((size_t)(n * 3 + c) << 10) + p] - 1.f;
    float q = rintf(h * 128.f);
    q = fminf(fmaxf(q, -128.f), 127.f);
    packed |= ((int)q & 0xff) << (8 * c);
  }
  ((int*)out)[i] = packed;
}

__global__ __launch_bounds__(256) void k_repack_conv(const float* __restrict__ w,
                                                     int* __restrict__ wp,
                                                     int Cin, int ntiles, int CB) {
  int total = 9 * CB * ntiles * 256;
  int i = blockIdx.x * 256 + threadIdx.x;
  if (i >= total) return;
  int v = i & 7, lane = (i >> 3) & 31;
  int t = i >> 8;
  int tile = t % ntiles, s = t / ntiles;
  int N = tile * 16 + (lane & 15);
  int kbase = ((v >> 2) * 32) + (lane >> 4) * 16 + (v & 3) * 4;
  int tap = s / CB, cb = s % CB;
  int packed = 0;
  for (int bb = 0; bb < 4; ++bb) {
    int cin = cb * 64 + kbase + bb;
    float wf = w[((size_t)N * Cin + cin) * 9 + tap];
    packed |= ((wf >= 0.f) ? 1 : 0xff) << (8 * bb);
  }
  wp[i] = packed;
}

__global__ __launch_bounds__(256) void k_repack_conv1(const float* __restrict__ w,
                                                      int* __restrict__ wp) {
  int i = blockIdx.x * 256 + threadIdx.x;
  if (i >= 4 * 256) return;
  int v = i & 7, lane = (i >> 3) & 31, tile = i >> 8;
  int N = tile * 16 + (lane & 15);
  int kbase = ((v >> 2) * 32) + (lane >> 4) * 16 + (v & 3) * 4;
  int packed = 0;
  for (int bb = 0; bb < 4; ++bb) {
    int K = kbase + bb;
    int tap = K >> 2, c = K & 3;
    int byte = 0;
    if (tap < 9 && c < 3) {
      float wf = w[((size_t)N * 3 + c) * 9 + tap];
      byte = (wf >= 0.f) ? 1 : 0xff;
    }
    packed |= byte << (8 * bb);
  }
  wp[i] = packed;
}

__global__ __launch_bounds__(256) void k_repack_fc(const float* __restrict__ w,
                                                   int* __restrict__ wp,
                                                   int Kdim, int Nout, int ntiles,
                                                   int perm) {
  int total = (Kdim >> 6) * ntiles * 256;
  int i = blockIdx.x * 256 + threadIdx.x;
  if (i >= total) return;
  int v = i & 7, lane = (i >> 3) & 31;
  int t = i >> 8;
  int tile = t % ntiles, s = t / ntiles;
  int N = tile * 16 + (lane & 15);
  int kbase = ((v >> 2) * 32) + (lane >> 4) * 16 + (v & 3) * 4;
  int packed = 0;
  for (int bb = 0; bb < 4; ++bb) {
    int K = s * 64 + kbase + bb;
    int col = perm ? ((K & 511) * 4 + (K >> 9)) : K;
    int byte = 0;
    if (N < Nout) {
      float wf = w[(size_t)N * Kdim + col];
      byte = (wf >= 0.f) ? 1 : 0xff;
    }
    packed |= byte << (8 * bb);
  }
  wp[i] = packed;
}

__device__ __forceinline__ void stage_conv_regs(
    const int8_t* __restrict__ in, int s, int CB, int layer1,
    int nS, int yS, int xS, int H, int W, int Cin, int half,
    int4& p0, int4& p1) {
  if (layer1) {
    int vals[8];
#pragma unroll
    for (int j = 0; j < 8; ++j) {
      int d = half * 8 + j;
      int dy = d / 3 - 1, dx = d % 3 - 1;
      int yy = yS + dy, xx = xS + dx;
      bool inb = (d < 9) && (yy >= 0) && (yy < H) && (xx >= 0) && (xx < W);
      int yc = min(max(yy, 0), H - 1), xc = min(max(xx, 0), W - 1);
      int v = ((const int*)in)[(size_t)(nS * H + yc) * W + xc];
      vals[j] = inb ? v : 0;
    }
    p0 = make_int4(vals[0], vals[1], vals[2], vals[3]);
    p1 = make_int4(vals[4], vals[5], vals[6], vals[7]);
  } else {
    int tap = s / CB, cb = s % CB;
    int dy = tap / 3 - 1, dx = tap % 3 - 1;
    int yy = yS + dy, xx = xS + dx;
    bool inb = (yy >= 0) && (yy < H) && (xx >= 0) && (xx < W);
    int yc = min(max(yy, 0), H - 1), xc = min(max(xx, 0), W - 1);
    const int4* src = (const int4*)(in +
        ((size_t)(nS * H + yc) * W + xc) * Cin + cb * 64 + half * 32);
    int4 a = src[0], b = src[1];
    p0 = inb ? a : make_int4(0, 0, 0, 0);
    p1 = inb ? b : make_int4(0, 0, 0, 0);
  }
}

__global__ __launch_bounds__(256) void k_bconv(
    const int8_t* __restrict__ in, const int* __restrict__ wp,
    int8_t* __restrict__ out,
    const float* __restrict__ bng, const float* __restrict__ bnb,
    const float* __restrict__ bnm, const float* __restrict__ bnv,
    int H, int W, int Cin, int Cout, int CB, int layer1, float scale) {
  __shared__ int smA[2][128 * 16];
  int tid = threadIdx.x, lane = tid & 31, wave = tid >> 5;
  int mBase = blockIdx.x * 128;
  int nTile = blockIdx.y;
  int ntiles16 = gridDim.y * 2;
  int steps = layer1 ? 1 : 9 * CB;

  int rowS = tid >> 1, half = tid & 1;
  int mS = mBase + rowS;
  int HW = H * W;
  int nS = mS / HW, remS = mS % HW, yS = remS / W, xS = remS % W;
  int sbase = rowS * 16 + half * 8;

  v8i acc0 = {0, 0, 0, 0, 0, 0, 0, 0};
  v8i acc1 = {0, 0, 0, 0, 0, 0, 0, 0};
  int rowA = wave * 16 + (lane & 15);
  int kadd = (lane >> 4) * 4;

  int4 p0, p1;
  stage_conv_regs(in, 0, CB, layer1, nS, yS, xS, H, W, Cin, half, p0, p1);
  smA[0][sbase + 0] = p0.x; smA[0][sbase + 1] = p0.y;
  smA[0][sbase + 2] = p0.z; smA[0][sbase + 3] = p0.w;
  smA[0][sbase + 4] = p1.x; smA[0][sbase + 5] = p1.y;
  smA[0][sbase + 6] = p1.z; smA[0][sbase + 7] = p1.w;

  for (int s = 0; s < steps; ++s) {
    __syncthreads();
    int cb_ = s & 1, nb = cb_ ^ 1;

    size_t wb = (((size_t)s * ntiles16 + nTile * 2) * 32 + lane) * 8;
    v8i bmat0 = *(const v8i*)(wp + wb);
    v8i bmat1 = *(const v8i*)(wp + wb + 256);

    v8i amat;
#pragma unroll
    for (int p = 0; p < 4; ++p) {
      int2 d2 = *(const int2*)&smA[cb_][rowA * 16 + (p >> 1) * 8 + (p & 1) * 2 + kadd];
      amat[2 * p] = d2.x;
      amat[2 * p + 1] = d2.y;
    }

    if (s + 1 < steps) {
      stage_conv_regs(in, s + 1, CB, layer1, nS, yS, xS, H, W, Cin, half, p0, p1);
      smA[nb][sbase + 0] = p0.x; smA[nb][sbase + 1] = p0.y;
      smA[nb][sbase + 2] = p0.z; smA[nb][sbase + 3] = p0.w;
      smA[nb][sbase + 4] = p1.x; smA[nb][sbase + 5] = p1.y;
      smA[nb][sbase + 6] = p1.z; smA[nb][sbase + 7] = p1.w;
    }

    acc0 = __builtin_amdgcn_wmma_i32_16x16x64_iu8(true, amat, true, bmat0, acc0,
                                                  false, false);
    acc1 = __builtin_amdgcn_wmma_i32_16x16x64_iu8(true, amat, true, bmat1, acc1,
                                                  false, false);
  }

  int ch0 = nTile * 32 + (lane & 15);
  int ch1 = ch0 + 16;
  float thr0 = bnm[ch0] - bnb[ch0] * sqrtf(bnv[ch0] + BN_EPS) / bng[ch0];
  float thr1 = bnm[ch1] - bnb[ch1] * sqrtf(bnv[ch1] + BN_EPS) / bng[ch1];
#pragma unroll
  for (int r = 0; r < 8; ++r) {
    int mm = mBase + wave * 16 + ((lane < 16) ? r : r + 8);
    float v0 = (float)acc0[r] * scale;
    float v1 = (float)acc1[r] * scale;
    out[(size_t)mm * Cout + ch0] = (v0 >= thr0) ? (int8_t)1 : (int8_t)-1;
    out[(size_t)mm * Cout + ch1] = (v1 >= thr1) ? (int8_t)1 : (int8_t)-1;
  }
}

__global__ __launch_bounds__(256) void k_pool(const int8_t* __restrict__ in,
                                              int8_t* __restrict__ out,
                                              int H, int W, int C) {
  int OH = H >> 1, OW = W >> 1, C4 = C >> 2;
  int total = 256 * OH * OW * C4;
  int i = blockIdx.x * 256 + threadIdx.x;
  if (i >= total) return;
  int cd = i % C4;
  int r = i / C4;
  int ox = r % OW; r /= OW;
  int oy = r % OH; int n = r / OH;
  const int* ip = (const int*)in;
  size_t b00 = ((size_t)(n * H + 2 * oy) * W + 2 * ox) * C4 + cd;
  size_t b10 = b00 + (size_t)W * C4;
  int a = ip[b00], b = ip[b00 + C4], c = ip[b10], d = ip[b10 + C4];
  int res = 0;
#pragma unroll
  for (int k = 0; k < 4; ++k) {
    int sh = 8 * k;
    int8_t m0 = (int8_t)(a >> sh), m1 = (int8_t)(b >> sh);
    int8_t m2 = (int8_t)(c >> sh), m3 = (int8_t)(d >> sh);
    int8_t mx = m0 > m1 ? m0 : m1;
    if (m2 > mx) mx = m2;
    if (m3 > mx) mx = m3;
    res |= ((int)(uint8_t)mx) << sh;
  }
  ((int*)out)[i] = res;
}

__global__ __launch_bounds__(256) void k_bfc(
    const int8_t* __restrict__ in, const int* __restrict__ wp,
    int8_t* __restrict__ out8, float* __restrict__ outf,
    const float* __restrict__ bng, const float* __restrict__ bnb,
    const float* __restrict__ bnm, const float* __restrict__ bnv,
    const float* __restrict__ tw, const float* __restrict__ tb,
    const float* __restrict__ tm, const float* __restrict__ tv,
    int K, int Nout, int finalLayer) {
  __shared__ int smA[2][128 * 16];
  int tid = threadIdx.x, lane = tid & 31, wave = tid >> 5;
  int mBase = blockIdx.x * 128;
  int nTile = blockIdx.y;
  int ntiles16 = gridDim.y * 2;
  int steps = K >> 6;
  int rowS = tid >> 1, half = tid & 1;
  int mS = mBase + rowS;
  int sbase = rowS * 16 + half * 8;

  v8i acc0 = {0, 0, 0, 0, 0, 0, 0, 0};
  v8i acc1 = {0, 0, 0, 0, 0, 0, 0, 0};
  int rowA = wave * 16 + (lane & 15);
  int kadd = (lane >> 4) * 4;

  const int4* srcRow = (const int4*)(in + (size_t)mS * K + half * 32);
  int4 p0 = srcRow[0], p1 = srcRow[1];
  smA[0][sbase + 0] = p0.x; smA[0][sbase + 1] = p0.y;
  smA[0][sbase + 2] = p0.z; smA[0][sbase + 3] = p0.w;
  smA[0][sbase + 4] = p1.x; smA[0][sbase + 5] = p1.y;
  smA[0][sbase + 6] = p1.z; smA[0][sbase + 7] = p1.w;

  for (int s = 0; s < steps; ++s) {
    __syncthreads();
    int cb_ = s & 1, nb = cb_ ^ 1;

    size_t wb = (((size_t)s * ntiles16 + nTile * 2) * 32 + lane) * 8;
    v8i bmat0 = *(const v8i*)(wp + wb);
    v8i bmat1 = *(const v8i*)(wp + wb + 256);

    v8i amat;
#pragma unroll
    for (int p = 0; p < 4; ++p) {
      int2 d2 = *(const int2*)&smA[cb_][rowA * 16 + (p >> 1) * 8 + (p & 1) * 2 + kadd];
      amat[2 * p] = d2.x;
      amat[2 * p + 1] = d2.y;
    }

    if (s + 1 < steps) {
      p0 = srcRow[(s + 1) * 4];
      p1 = srcRow[(s + 1) * 4 + 1];
      smA[nb][sbase + 0] = p0.x; smA[nb][sbase + 1] = p0.y;
      smA[nb][sbase + 2] = p0.z; smA[nb][sbase + 3] = p0.w;
      smA[nb][sbase + 4] = p1.x; smA[nb][sbase + 5] = p1.y;
      smA[nb][sbase + 6] = p1.z; smA[nb][sbase + 7] = p1.w;
    }

    acc0 = __builtin_amdgcn_wmma_i32_16x16x64_iu8(true, amat, true, bmat0, acc0,
                                                  false, false);
    acc1 = __builtin_amdgcn_wmma_i32_16x16x64_iu8(true, amat, true, bmat1, acc1,
                                                  false, false);
  }

  int ch0 = nTile * 32 + (lane & 15);
  int ch1 = ch0 + 16;
  if (finalLayer) {
    float inv = rsqrtf(tv[0] + BN_EPS);
    float w0 = tw[0], b0 = tb[0], m0 = tm[0];
#pragma unroll
    for (int r = 0; r < 8; ++r) {
      int mm = mBase + wave * 16 + ((lane < 16) ? r : r + 8);
      if (ch0 < Nout)
        outf[(size_t)mm * Nout + ch0] = ((float)acc0[r] - m0) * inv * w0 + b0;
      if (ch1 < Nout)
        outf[(size_t)mm * Nout + ch1] = ((float)acc1[r] - m0) * inv * w0 + b0;
    }
  } else {
    float thr0 = bnm[ch0] - bnb[ch0] * sqrtf(bnv[ch0] + BN_EPS) / bng[ch0];
    float thr1 = bnm[ch1] - bnb[ch1] * sqrtf(bnv[ch1] + BN_EPS) / bng[ch1];
#pragma unroll
    for (int r = 0; r < 8; ++r) {
      int mm = mBase + wave * 16 + ((lane < 16) ? r : r + 8);
      out8[(size_t)mm * Nout + ch0] =
          ((float)acc0[r] >= thr0) ? (int8_t)1 : (int8_t)-1;
      out8[(size_t)mm * Nout + ch1] =
          ((float)acc1[r] >= thr1) ? (int8_t)1 : (int8_t)-1;
    }
  }
}

extern "C" void kernel_launch(void* const* d_in, const int* in_sizes, int n_in,
                              void* d_out, int out_size, void* d_ws,
                              size_t ws_size, hipStream_t stream) {
  (void)in_sizes; (void)n_in; (void)out_size; (void)ws_size;
  const float* x = (const float*)d_in[0];
  const float* convw[8];
  for (int i = 0; i < 8; ++i) convw[i] = (const float*)d_in[1 + i];
  const float* bn2d[8][4];
  for (int i = 0; i < 8; ++i)
    for (int j = 0; j < 4; ++j) bn2d[i][j] = (const float*)d_in[9 + 4 * i + j];
  const float* fcw[3];
  for (int i = 0; i < 3; ++i) fcw[i] = (const float*)d_in[41 + i];
  const float* bn1d[2][4];
  for (int i = 0; i < 2; ++i)
    for (int j = 0; j < 4; ++j) bn1d[i][j] = (const float*)d_in[44 + 4 * i + j];
  const float* tnw = (const float*)d_in[52];
  const float* tnb = (const float*)d_in[53];
  const float* tnm = (const float*)d_in[54];
  const float* tnv = (const float*)d_in[55];

  uint8_t* ws = (uint8_t*)d_ws;
  int8_t* q    = (int8_t*)ws;
  int8_t* bufA = (int8_t*)(ws + (1u << 20));
  int8_t* bufB = (int8_t*)(ws + (1u << 20) + (17u << 20));
  int*    wp   = (int*)(ws + (1u << 20) + (34u << 20));

  k_quant<<<1024, 256, 0, stream>>>(x, q);

  k_repack_conv1<<<4, 256, 0, stream>>>(convw[0], wp);
  k_bconv<<<dim3(2048, 2), 256, 0, stream>>>(
      q, wp, bufA, bn2d[0][0], bn2d[0][1], bn2d[0][2], bn2d[0][3],
      32, 32, 4, 64, 1, 1, 1.f / 128.f);

  const int Hs[8]    = {32, 32, 16, 16, 8, 8, 4, 4};
  const int Cins[8]  = {4, 64, 64, 128, 128, 256, 256, 512};
  const int Couts[8] = {64, 64, 128, 128, 256, 256, 512, 512};
  const int pools[8] = {0, 1, 0, 1, 0, 1, 0, 1};

  int8_t* cur = bufA;
  int8_t* nxt = bufB;
  for (int L = 1; L < 8; ++L) {
    int H = Hs[L], W = H, Cin = Cins[L], Cout = Couts[L], CB = Cin / 64;
    int ntiles = Cout / 16;
    int rpTotal = 9 * CB * ntiles * 256;
    k_repack_conv<<<(rpTotal + 255) / 256, 256, 0, stream>>>(convw[L], wp, Cin,
                                                             ntiles, CB);
    int M = 256 * H * W;
    k_bconv<<<dim3(M / 128, Cout / 32), 256, 0, stream>>>(
        cur, wp, nxt, bn2d[L][0], bn2d[L][1], bn2d[L][2], bn2d[L][3],
        H, W, Cin, Cout, CB, 0, 1.f);
    { int8_t* t = cur; cur = nxt; nxt = t; }
    if (pools[L]) {
      int pTotal = 256 * (H / 2) * (W / 2) * (Cout / 4);
      k_pool<<<(pTotal + 255) / 256, 256, 0, stream>>>(cur, nxt, H, W, Cout);
      int8_t* t = cur; cur = nxt; nxt = t;
    }
  }

  {
    int Kd = 2048, Nout = 1024, nt = 64;
    int rp = (Kd / 64) * nt * 256;
    k_repack_fc<<<(rp + 255) / 256, 256, 0, stream>>>(fcw[0], wp, Kd, Nout, nt, 1);
    k_bfc<<<dim3(2, nt / 2), 256, 0, stream>>>(
        cur, wp, nxt, nullptr, bn1d[0][0], bn1d[0][1], bn1d[0][2], bn1d[0][3],
        nullptr, nullptr, nullptr, nullptr, Kd, Nout, 0);
    int8_t* t = cur; cur = nxt; nxt = t;
  }
  {
    int Kd = 1024, Nout = 512, nt = 32;
    int rp = (Kd / 64) * nt * 256;
    k_repack_fc<<<(rp + 255) / 256, 256, 0, stream>>>(fcw[1], wp, Kd, Nout, nt, 0);
    k_bfc<<<dim3(2, nt / 2), 256, 0, stream>>>(
        cur, wp, nxt, nullptr, bn1d[1][0], bn1d[1][1], bn1d[1][2], bn1d[1][3],
        nullptr, nullptr, nullptr, nullptr, Kd, Nout, 0);
    int8_t* t = cur; cur = nxt; nxt = t;
  }
  {
    int Kd = 512, Nout = 200, nt = 14;
    int rp = (Kd / 64) * nt * 256;
    k_repack_fc<<<(rp + 255) / 256, 256, 0, stream>>>(fcw[2], wp, Kd, Nout, nt, 0);
    k_bfc<<<dim3(2, nt / 2), 256, 0, stream>>>(
        cur, wp, nullptr, (float*)d_out, nullptr, nullptr, nullptr, nullptr,
        tnw, tnb, tnm, tnv, Kd, Nout, 1);
  }
}
